// QuantumLM_57904749085282
// MI455X (gfx1250) — matrix operand order repaired and checked by reading, not compiled, not run on hardware
//
#include <hip/hip_runtime.h>
#include <hip/hip_bf16.h>

// Problem constants (must match reference)
constexpr int B_ = 1024, T_ = 64, V_ = 1024, E_ = 512, M_ = 1024, L_ = 32;
constexpr int PE_ = M_ / 2;        // 512 even pairs
constexpr int PO_ = (M_ - 1) / 2;  // 511 odd pairs

typedef __attribute__((ext_vector_type(16))) __bf16 v16bf;
typedef __attribute__((ext_vector_type(8)))  float  v8f;
typedef __attribute__((ext_vector_type(4)))  unsigned int u32x4;
typedef __attribute__((ext_vector_type(2)))  int i32x2;
typedef __attribute__((ext_vector_type(4)))  int i32x4;

union ABFrag { v16bf v; u32x4 q[2]; };

// ---------------------------------------------------------------------------
// CDNA5 async global->LDS copy helpers (ASYNCcnt-tracked)
// Builtin prototype (probe-confirmed via hipcc diagnostic):
//   (global int-vector*, lds int-vector*, imm offset, imm cpol)
// ---------------------------------------------------------------------------
__device__ __forceinline__ void async_copy_b64(const void* g, void* l) {
#if __has_builtin(__builtin_amdgcn_global_load_async_to_lds_b64)
  __builtin_amdgcn_global_load_async_to_lds_b64(
      (__attribute__((address_space(1))) i32x2*)g,
      (__attribute__((address_space(3))) i32x2*)l, 0, 0);
#else
  asm volatile("global_load_async_to_lds_b64 %0, %1, off"
               :: "v"((unsigned)(unsigned long long)l), "v"(g) : "memory");
#endif
}

__device__ __forceinline__ void async_copy_b128(const void* g, void* l) {
#if __has_builtin(__builtin_amdgcn_global_load_async_to_lds_b128)
  __builtin_amdgcn_global_load_async_to_lds_b128(
      (__attribute__((address_space(1))) i32x4*)g,
      (__attribute__((address_space(3))) i32x4*)l, 0, 0);
#else
  asm volatile("global_load_async_to_lds_b128 %0, %1, off"
               :: "v"((unsigned)(unsigned long long)l), "v"(g) : "memory");
#endif
}

__device__ __forceinline__ void wait_async0() {
#if __has_builtin(__builtin_amdgcn_s_wait_asynccnt)
  __builtin_amdgcn_s_wait_asynccnt(0);
#else
  asm volatile("s_wait_asynccnt 0x0" ::: "memory");
#endif
}

// ---------------------------------------------------------------------------
// Small prep kernels
// ---------------------------------------------------------------------------
__global__ void cvt_bf16_kernel(const float* __restrict__ src,
                                __hip_bfloat16* __restrict__ dst, int n) {
  int i = blockIdx.x * blockDim.x + threadIdx.x;
  if (i < n) dst[i] = __float2bfloat16(src[i]);
}

// Rmat is (2V, 2M) (N,K)-major:
//   rows 0..V-1   (real cols):  [ Rre | -Rim ]
//   rows V..2V-1  (imag cols):  [ Rim |  Rre ]
__global__ void build_rmat_kernel(const float* __restrict__ Rre,
                                  const float* __restrict__ Rim,
                                  __hip_bfloat16* __restrict__ Rmat) {
  int idx = blockIdx.x * blockDim.x + threadIdx.x;  // over 2V*2M
  int n = idx / (2 * M_);
  int k = idx - n * (2 * M_);
  float val;
  if (n < V_) {
    val = (k < M_) ? Rre[n * M_ + k] : -Rim[n * M_ + (k - M_)];
  } else {
    int v = n - V_;
    val = (k < M_) ? Rim[v * M_ + k] : Rre[v * M_ + (k - M_)];
  }
  Rmat[idx] = __float2bfloat16(val);
}

__global__ void init_h_kernel(float* __restrict__ h,
                              __hip_bfloat16* __restrict__ hbf) {
  int i = blockIdx.x * blockDim.x + threadIdx.x;
  if (i < B_ * E_) { h[i] = 0.f; hbf[i] = __float2bfloat16(0.f); }
}

// ---------------------------------------------------------------------------
// bf16 WMMA GEMM with double-buffered async-LDS pipeline:
//   C[m,n] = sum_k A[m,k]*Bt[n,k] (+ bias[n])
//   A : (Mrows, K) bf16 row-major; optional row gather (embedding lookup)
//   Bt: (N, K)     bf16 row-major (B transposed -> contiguous K per column)
//   C : (Mrows, N) f32 or bf16
// Block: 256 threads = 8 waves (2x4); block tile 32x256; wave tile 16x64.
// Per K-step (32): stage A(32x32) + B(256x32) tiles into LDS with
// global_load_async_to_lds (ASYNCcnt). All fragments are pulled from LDS
// (ds_load_b128) BEFORE the 4 WMMAs so a single s_wait_dscnt covers them and
// the WMMAs issue back-to-back into the matrix pipe.
// ---------------------------------------------------------------------------
__global__ __launch_bounds__(256)
void gemm_bf16_wmma(const __hip_bfloat16* __restrict__ A,
                    const int* __restrict__ gather,
                    const __hip_bfloat16* __restrict__ Bt,
                    const float* __restrict__ bias,
                    void* __restrict__ C, int c_is_bf16,
                    int Mrows, int N, int K) {
  __shared__ unsigned short sA[2][32 * 32];    //  2 x 2 KB
  __shared__ unsigned short sB[2][256 * 32];   //  2 x 16 KB

  const int tid  = threadIdx.x;
  const int lane = tid & 31;
  const int wave = tid >> 5;
  const int wrow = wave >> 2;   // 0..1
  const int wcol = wave & 3;    // 0..3
  const int r    = lane & 15;
  const int sel  = lane >> 4;   // half-wave select

  const int mrowB = blockIdx.y * 32;
  const int ncolB = blockIdx.x * 256;

  // -- per-thread async-copy assignments --
  // A tile: 32 rows x 32 K = 256 chunks of 8B; thread tid -> chunk tid
  const int ar = tid >> 3;            // 0..31 local row
  const int ak = (tid & 7) * 4;       // K element offset
  int agrow = mrowB + ar;
  if (gather) agrow = gather[agrow];
  const __hip_bfloat16* aSrc = A + (size_t)agrow * K + ak;
  const int aOff = ar * 32 + ak;

  // B tile: 256 cols x 32 K = 1024 chunks of 16B; thread -> chunks tid+256*j
  const __hip_bfloat16* bSrc[4];
  int bOff[4];
#pragma unroll
  for (int j = 0; j < 4; ++j) {
    int c = tid + 256 * j;
    int n = c >> 2;
    int k = (c & 3) * 8;
    bSrc[j] = Bt + (size_t)(ncolB + n) * K + k;
    bOff[j] = n * 32 + k;
  }

  // -- fragment LDS read addresses (ISA 16-bit A/B WMMA layouts) --
  const int aFragOff = (wrow * 16 + r) * 32 + sel * 8;   // K = sel*8.., +16..
  int bFragOff[4];
#pragma unroll
  for (int s = 0; s < 4; ++s)
    bFragOff[s] = (wcol * 64 + s * 16 + r) * 32 + sel * 8;  // K = sel*8.., +16..

  v8f acc[4] = {};
  const int nIt = K >> 5;

  // prologue: stage 0 -> buffer 0
  async_copy_b64(aSrc, &sA[0][aOff]);
#pragma unroll
  for (int j = 0; j < 4; ++j) async_copy_b128(bSrc[j], &sB[0][bOff[j]]);

  for (int it = 0; it < nIt; ++it) {
    wait_async0();        // stage `it` copies (this wave's) complete
    __syncthreads();      // all waves' copies visible; prev buffer free
    const int p = it & 1;
    if (it + 1 < nIt) {   // stage `it+1` -> other buffer, overlaps WMMAs below
      const int k0n = (it + 1) << 5;
      async_copy_b64(aSrc + k0n, &sA[p ^ 1][aOff]);
#pragma unroll
      for (int j = 0; j < 4; ++j)
        async_copy_b128(bSrc[j] + k0n, &sB[p ^ 1][bOff[j]]);
    }
    // Load A fragment + all 4 B fragments first (one dscnt wait), then
    // issue the 4 independent WMMAs back-to-back.
    ABFrag a, bfr[4];
    const unsigned short* af = &sA[p][aFragOff];
    a.q[0] = *reinterpret_cast<const u32x4*>(af);
    a.q[1] = *reinterpret_cast<const u32x4*>(af + 16);
#pragma unroll
    for (int s = 0; s < 4; ++s) {
      const unsigned short* bf = &sB[p][bFragOff[s]];
      bfr[s].q[0] = *reinterpret_cast<const u32x4*>(bf);
      bfr[s].q[1] = *reinterpret_cast<const u32x4*>(bf + 16);
    }
#pragma unroll
    for (int s = 0; s < 4; ++s)
      acc[s] = __builtin_amdgcn_wmma_f32_16x16x32_bf16(
          /*neg_a=*/false, a.v, /*neg_b=*/false, bfr[s].v,
          /*c_mod=*/(short)0, acc[s], /*reuse_a=*/false, /*reuse_b=*/false);
  }

  // 16x16 f32 C layout: lanes 0-15 -> M = 0..7, lanes 16-31 -> M = 8..15
  const int orow0 = mrowB + wrow * 16 + sel * 8;
  const int ncol0 = ncolB + wcol * 64;
#pragma unroll
  for (int s = 0; s < 4; ++s) {
    const int ocol = ncol0 + s * 16 + r;
    const float bv = bias ? bias[ocol] : 0.f;
    if (c_is_bf16) {
      __hip_bfloat16* Cb = (__hip_bfloat16*)C;
#pragma unroll
      for (int i = 0; i < 8; ++i)
        Cb[(size_t)(orow0 + i) * N + ocol] = __float2bfloat16(acc[s][i] + bv);
    } else {
      float* Cf = (float*)C;
#pragma unroll
      for (int i = 0; i < 8; ++i)
        Cf[(size_t)(orow0 + i) * N + ocol] = acc[s][i] + bv;
    }
  }
}

// ---------------------------------------------------------------------------
// GRU gate update (torch gate order r,z,n)
// ---------------------------------------------------------------------------
__device__ __forceinline__ float sigm(float x) { return 1.f / (1.f + expf(-x)); }

__global__ void gru_update_kernel(const __hip_bfloat16* __restrict__ xg, // (B,T,3E)
                                  const float* __restrict__ hg,          // (B,3E)
                                  float* __restrict__ h,                 // (B,E)
                                  __hip_bfloat16* __restrict__ hbf, int t) {
  int idx = blockIdx.x * blockDim.x + threadIdx.x;
  if (idx >= B_ * E_) return;
  int b = idx >> 9;       // / 512
  int e = idx & 511;
  size_t xrow = ((size_t)b * T_ + t) * (3 * E_);
  float ir  = __bfloat162float(xg[xrow + e]);
  float iz  = __bfloat162float(xg[xrow + E_ + e]);
  float in_ = __bfloat162float(xg[xrow + 2 * E_ + e]);
  size_t hrow = (size_t)b * (3 * E_);
  float hr = hg[hrow + e], hz = hg[hrow + E_ + e], hn = hg[hrow + 2 * E_ + e];
  float rg = sigm(ir + hr);
  float zg = sigm(iz + hz);
  float ng = tanhf(in_ + rg * hn);
  float hv = h[idx];
  float hnew = (1.f - zg) * ng + zg * hv;
  h[idx] = hnew;
  hbf[idx] = __float2bfloat16(hnew);
}

// ---------------------------------------------------------------------------
// MZI quantum core: one 256-thread block per batch row, state in LDS
// ---------------------------------------------------------------------------
__device__ float block_sum(float v, float* red) {
  int tid = threadIdx.x;
  red[tid] = v;
  __syncthreads();
  for (int s = 128; s > 0; s >>= 1) {
    if (tid < s) red[tid] += red[tid + s];
    __syncthreads();
  }
  float t = red[0];
  __syncthreads();
  return t;
}

__device__ __forceinline__ void mzi_pair(float* sre, float* sim, int i0, int i1,
                                         float th, float ph, float rh) {
  float st, ct; sincosf(th, &st, &ct);
  float sp, cp; sincosf(ph, &sp, &cp);
  float sr, cr; sincosf(rh, &sr, &cr);
  // u11 = c*e^{i(p+r)}, u12 = i*s*e^{ip}, u21 = i*s*e^{ir}, u22 = c
  float cpr = cp * cr - sp * sr, spr = sp * cr + cp * sr;
  float u11r = ct * cpr, u11i = ct * spr;
  float u12r = -st * sp, u12i = st * cp;
  float u21r = -st * sr, u21i = st * cr;
  float u22r = ct;
  float v0r = sre[i0], v0i = sim[i0];
  float v1r = sre[i1], v1i = sim[i1];
  float n0r = u11r * v0r - u11i * v0i + u12r * v1r - u12i * v1i;
  float n0i = u11r * v0i + u11i * v0r + u12r * v1i + u12i * v1r;
  float n1r = u21r * v0r - u21i * v0i + u22r * v1r;
  float n1i = u21r * v0i + u21i * v0r + u22r * v1i;
  sre[i0] = n0r; sim[i0] = n0i;
  sre[i1] = n1r; sim[i1] = n1i;
}

__global__ __launch_bounds__(256)
void mzi_kernel(const float* __restrict__ zc,     // (B, 2M) [re | im]
                const float* __restrict__ phase,  // (L, M)
                const float* __restrict__ te, const float* __restrict__ pe,
                const float* __restrict__ re_,
                const float* __restrict__ to, const float* __restrict__ po,
                const float* __restrict__ ro,
                __hip_bfloat16* __restrict__ psi_pack) {  // (B, 2M) [re | im]
  __shared__ float sre[M_];
  __shared__ float sim[M_];
  __shared__ float red[256];
  const int b = blockIdx.x, tid = threadIdx.x;

  float acc = 0.f;
  for (int j = tid; j < M_; j += 256) {
    float rr = zc[(size_t)b * (2 * M_) + j];
    float ii = zc[(size_t)b * (2 * M_) + M_ + j];
    sre[j] = rr; sim[j] = ii;
    acc += rr * rr + ii * ii;
  }
  float tot = block_sum(acc, red);
  float inv = rsqrtf(tot + 1e-8f);
  for (int j = tid; j < M_; j += 256) { sre[j] *= inv; sim[j] *= inv; }
  __syncthreads();

  for (int l = 0; l < L_; ++l) {
    // per-mode phase
    for (int j = tid; j < M_; j += 256) {
      float sp, cp; sincosf(phase[l * M_ + j], &sp, &cp);
      float rr = sre[j], ii = sim[j];
      sre[j] = rr * cp - ii * sp;
      sim[j] = rr * sp + ii * cp;
    }
    __syncthreads();
    // even pairs (0,1),(2,3),...
    for (int p = tid; p < PE_; p += 256)
      mzi_pair(sre, sim, 2 * p, 2 * p + 1,
               te[l * PE_ + p], pe[l * PE_ + p], re_[l * PE_ + p]);
    __syncthreads();
    // odd pairs (1,2),(3,4),...
    for (int p = tid; p < PO_; p += 256)
      mzi_pair(sre, sim, 2 * p + 1, 2 * p + 2,
               to[l * PO_ + p], po[l * PO_ + p], ro[l * PO_ + p]);
    __syncthreads();
    // renormalize
    float a2 = 0.f;
    for (int j = tid; j < M_; j += 256) a2 += sre[j] * sre[j] + sim[j] * sim[j];
    tot = block_sum(a2, red);
    inv = rsqrtf(tot + 1e-8f);
    for (int j = tid; j < M_; j += 256) { sre[j] *= inv; sim[j] *= inv; }
    __syncthreads();
  }

  for (int j = tid; j < M_; j += 256) {
    psi_pack[(size_t)b * (2 * M_) + j]      = __float2bfloat16(sre[j]);
    psi_pack[(size_t)b * (2 * M_) + M_ + j] = __float2bfloat16(sim[j]);
  }
}

// ---------------------------------------------------------------------------
// Born-rule log-prob + logsumexp: one block per batch row
// ---------------------------------------------------------------------------
__global__ __launch_bounds__(256)
void logprob_kernel(const float* __restrict__ amps,  // (B, 2V) [re | im]
                    float* __restrict__ out) {       // (B, V)
  __shared__ float lm[V_];
  __shared__ float red[256];
  const int b = blockIdx.x, tid = threadIdx.x;

  float mx = -3.4e38f;
  for (int v = tid; v < V_; v += 256) {
    float rr = amps[(size_t)b * (2 * V_) + v];
    float ii = amps[(size_t)b * (2 * V_) + V_ + v];
    float l = logf(rr * rr + ii * ii + 1e-12f);
    lm[v] = l;
    mx = fmaxf(mx, l);
  }
  red[tid] = mx;
  __syncthreads();
  for (int s = 128; s > 0; s >>= 1) {
    if (tid < s) red[tid] = fmaxf(red[tid], red[tid + s]);
    __syncthreads();
  }
  mx = red[0];
  __syncthreads();

  float sum = 0.f;
  for (int v = tid; v < V_; v += 256) sum += expf(lm[v] - mx);
  float tot = block_sum(sum, red);
  float lse = mx + logf(tot);
  for (int v = tid; v < V_; v += 256) out[(size_t)b * V_ + v] = lm[v] - lse;
}

// ---------------------------------------------------------------------------
// Host orchestration
// ---------------------------------------------------------------------------
extern "C" void kernel_launch(void* const* d_in, const int* in_sizes, int n_in,
                              void* d_out, int out_size, void* d_ws, size_t ws_size,
                              hipStream_t stream) {
  (void)in_sizes; (void)n_in; (void)out_size; (void)ws_size;

  const int*   tokens = (const int*)  d_in[0];
  const float* embed  = (const float*)d_in[1];
  const float* W_ih   = (const float*)d_in[2];
  const float* W_hh   = (const float*)d_in[3];
  const float* b_ih   = (const float*)d_in[4];
  const float* b_hh   = (const float*)d_in[5];
  const float* Wc     = (const float*)d_in[6];
  const float* bc     = (const float*)d_in[7];
  const float* phase  = (const float*)d_in[8];
  const float* te     = (const float*)d_in[9];
  const float* pe     = (const float*)d_in[10];
  const float* re_    = (const float*)d_in[11];
  const float* to     = (const float*)d_in[12];
  const float* po     = (const float*)d_in[13];
  const float* ro     = (const float*)d_in[14];
  const float* Rre    = (const float*)d_in[15];
  const float* Rim    = (const float*)d_in[16];
  float* out = (float*)d_out;

  char* ws = (char*)d_ws;
  size_t off = 0;
  auto take = [&](size_t bytes) -> char* {
    char* p = ws + off;
    off = (off + bytes + 255) & ~(size_t)255;
    return p;
  };

  __hip_bfloat16* embed_bf = (__hip_bfloat16*)take((size_t)V_ * E_ * 2);
  __hip_bfloat16* Bih      = (__hip_bfloat16*)take((size_t)3 * E_ * E_ * 2);
  __hip_bfloat16* Bhh      = (__hip_bfloat16*)take((size_t)3 * E_ * E_ * 2);
  __hip_bfloat16* Bwc      = (__hip_bfloat16*)take((size_t)2 * M_ * E_ * 2);
  __hip_bfloat16* Rmat     = (__hip_bfloat16*)take((size_t)(2 * V_) * (2 * M_) * 2);
  __hip_bfloat16* xg       = (__hip_bfloat16*)take((size_t)B_ * T_ * 3 * E_ * 2);
  float*          h        = (float*)         take((size_t)B_ * E_ * 4);
  __hip_bfloat16* hbf      = (__hip_bfloat16*)take((size_t)B_ * E_ * 2);
  float*          hg       = (float*)         take((size_t)B_ * 3 * E_ * 4);
  float*          zc       = (float*)         take((size_t)B_ * 2 * M_ * 4);
  __hip_bfloat16* psi      = (__hip_bfloat16*)take((size_t)B_ * 2 * M_ * 2);
  float*          amps     = (float*)         take((size_t)B_ * 2 * V_ * 4);

  int n;
  n = V_ * E_;
  cvt_bf16_kernel<<<(n + 255) / 256, 256, 0, stream>>>(embed, embed_bf, n);
  n = 3 * E_ * E_;
  cvt_bf16_kernel<<<(n + 255) / 256, 256, 0, stream>>>(W_ih, Bih, n);
  cvt_bf16_kernel<<<(n + 255) / 256, 256, 0, stream>>>(W_hh, Bhh, n);
  n = 2 * M_ * E_;
  cvt_bf16_kernel<<<(n + 255) / 256, 256, 0, stream>>>(Wc, Bwc, n);
  n = (2 * V_) * (2 * M_);
  build_rmat_kernel<<<(n + 255) / 256, 256, 0, stream>>>(Rre, Rim, Rmat);
  n = B_ * E_;
  init_h_kernel<<<(n + 255) / 256, 256, 0, stream>>>(h, hbf);

  // x_gates = embed[tokens] @ W_ih^T + b_ih   (gathered A, bf16 output)
  gemm_bf16_wmma<<<dim3((3 * E_) / 256, (B_ * T_) / 32), 256, 0, stream>>>(
      embed_bf, tokens, Bih, b_ih, (void*)xg, 1, B_ * T_, 3 * E_, E_);

  // GRU recurrence
  for (int t = 0; t < T_; ++t) {
    gemm_bf16_wmma<<<dim3((3 * E_) / 256, B_ / 32), 256, 0, stream>>>(
        hbf, nullptr, Bhh, b_hh, (void*)hg, 0, B_, 3 * E_, E_);
    gru_update_kernel<<<(B_ * E_ + 255) / 256, 256, 0, stream>>>(xg, hg, h, hbf, t);
  }

  // zc = h @ Wc^T + bc
  gemm_bf16_wmma<<<dim3((2 * M_) / 256, B_ / 32), 256, 0, stream>>>(
      hbf, nullptr, Bwc, bc, (void*)zc, 0, B_, 2 * M_, E_);

  // quantum core -> packed [re|im] bf16 psi
  mzi_kernel<<<B_, 256, 0, stream>>>(zc, phase, te, pe, re_, to, po, ro, psi);

  // amps = psi @ Rmat^T  (complex GEMM packed as one real GEMM)
  gemm_bf16_wmma<<<dim3((2 * V_) / 256, B_ / 32), 256, 0, stream>>>(
      psi, nullptr, Rmat, nullptr, (void*)amps, 0, B_, 2 * V_, 2 * M_);

  // Born-rule log-probs with logsumexp
  logprob_kernel<<<B_, 256, 0, stream>>>(amps, out);
}
